// MultiHeadAttentionAttentionPooling_35442070127371
// MI455X (gfx1250) — matrix operand order repaired and checked by reading, not compiled, not run on hardware
//
#include <hip/hip_runtime.h>
#include <cmath>

// ---------------------------------------------------------------------------
// Types for CDNA5 WMMA (gfx1250, wave32)
// ---------------------------------------------------------------------------
typedef __bf16 bf16;
typedef __bf16 v16bf __attribute__((ext_vector_type(16)));
typedef float  v8f   __attribute__((ext_vector_type(8)));

union FragB16 { v16bf v; unsigned int u[8]; };

// Tile configuration: 256 threads = 8 waves; block tile 128x128, K-step 32.
// Waves arranged 2x4; each wave owns a 64x32 sub-tile = 4x2 WMMA 16x16 tiles.
static constexpr int BM = 128;
static constexpr int BN = 128;
static constexpr int BK = 32;
static constexpr int LDA_S  = 40;       // As row stride (elems): 80B rows -> 16B aligned
static constexpr int LDBT_S = 40;       // Bs row stride for B^T layout
static constexpr int LDBN_S = BN + 8;   // Bs row stride for natural [K][N] layout (272B)

// C = alpha * (A @ B) + bias  (BT=false: B is [K,N] row-major)
// C = alpha * (A @ B^T)       (BT=true : B is [N,K] row-major)
// Writes f32 to Cf if non-null, else bf16 to Ch. blockIdx.z batches with
// element strides sAz/sBz/sCz. Double-buffered LDS, one barrier per K tile.
template<bool BT>
__global__ __launch_bounds__(256)
void gemm_wmma(const bf16* __restrict__ A, int lda, long sAz,
               const bf16* __restrict__ B, int ldb, long sBz,
               float* __restrict__ Cf, bf16* __restrict__ Ch, int ldc, long sCz,
               int K, float alpha, const float* __restrict__ bias)
{
    __shared__ __align__(16) bf16 As[2][BM][LDA_S];
    __shared__ __align__(16) bf16 Bs[2][BT ? BN : BK][BT ? LDBT_S : LDBN_S];

    const int tid  = threadIdx.x;
    const int lane = tid & 31;
    const int hl   = lane >> 4;   // half-wave: 0 or 1
    const int l    = lane & 15;
    const int wave = tid >> 5;    // 0..7
    const int wm   = wave >> 2;   // 0..1  (row group of 64)
    const int wn   = wave & 3;    // 0..3  (col group of 32)

    const long z  = blockIdx.z;
    const int  bm = blockIdx.y * BM;
    const int  bn = blockIdx.x * BN;

    const bf16* Ab = A + z * sAz + (long)bm * lda;
    const bf16* Bb = BT ? (B + z * sBz + (long)bn * ldb)
                        : (B + z * sBz + (long)bn);

    // per-thread tile-loader coordinates
    const int ra = tid >> 2, sa = tid & 3;    // A (and B^T) tiles: 128 rows x 64B
    const int rb = tid >> 4, sb = tid & 15;   // NN B tile: 32 rows x 256B

    // strength-reduced global pointers (advance by constants each K step)
    const bf16* aP0 = Ab + (long)ra * lda + sa * 8;
    const bf16* aP1 = Ab + (long)(ra + 64) * lda + sa * 8;
    const bf16* bP0;
    const bf16* bP1;
    long bStep;
    if (BT) {
        bP0 = Bb + (long)ra * ldb + sa * 8;
        bP1 = Bb + (long)(ra + 64) * ldb + sa * 8;
        bStep = BK;
    } else {
        bP0 = Bb + (long)rb * ldb + sb * 8;
        bP1 = Bb + (long)(rb + 16) * ldb + sb * 8;
        bStep = (long)BK * ldb;
    }

    v8f acc[4][2];
#pragma unroll
    for (int mi = 0; mi < 4; ++mi)
#pragma unroll
        for (int ni = 0; ni < 2; ++ni)
#pragma unroll
            for (int r = 0; r < 8; ++r) acc[mi][ni][r] = 0.0f;

    // ---- prologue: load tile 0 and stage into LDS buffer 0
    uint4 a0 = *(const uint4*)aP0;
    uint4 a1 = *(const uint4*)aP1;
    uint4 b0 = *(const uint4*)bP0;
    uint4 b1 = *(const uint4*)bP1;
    *(uint4*)&As[0][ra][sa * 8]      = a0;
    *(uint4*)&As[0][ra + 64][sa * 8] = a1;
    if (BT) {
        *(uint4*)&Bs[0][ra][sa * 8]      = b0;
        *(uint4*)&Bs[0][ra + 64][sa * 8] = b1;
    } else {
        *(uint4*)&Bs[0][rb][sb * 8]      = b0;
        *(uint4*)&Bs[0][rb + 16][sb * 8] = b1;
    }
    __syncthreads();

    int buf = 0;
    for (int k0 = 0;; k0 += BK) {
        const bool hasNext = (k0 + BK) < K;
        if (hasNext) {
            // issue next tile's global loads; they overlap the WMMAs below
            aP0 += BK; aP1 += BK; bP0 += bStep; bP1 += bStep;
            a0 = *(const uint4*)aP0;
            a1 = *(const uint4*)aP1;
            b0 = *(const uint4*)bP0;
            b1 = *(const uint4*)bP1;
            // prefetch the tile after next (gfx1250 global_prefetch_b8)
            __builtin_prefetch((const void*)(aP0 + BK), 0, 0);
            __builtin_prefetch((const void*)(bP0 + bStep), 0, 0);
        }

        // ---- assemble fragments from LDS[buf] in this code's contiguous k order (k = 16*half + s), A and B alike
        // A 16x32 bf16: lanes 0-15 = M0-15 {K0..15}, lanes 16-31 = {K16..31}
        FragB16 af[4];
#pragma unroll
        for (int mi = 0; mi < 4; ++mi) {
            const unsigned int* Ar =
                (const unsigned int*)&As[buf][wm * 64 + mi * 16 + l][0];
#pragma unroll
            for (int j = 0; j < 4; ++j) af[mi].u[j]     = Ar[8 * hl + j];
#pragma unroll
            for (int j = 0; j < 4; ++j) af[mi].u[4 + j] = Ar[8 * hl + 4 + j];
        }
        // B 32x16 bf16: VGPR r lanes 0-15 = K{2r,2r+1}, lanes 16-31 = K{16+2r,17+2r}
        FragB16 bfr[2];
#pragma unroll
        for (int ni = 0; ni < 2; ++ni) {
            if (BT) {
                const unsigned int* Br =
                    (const unsigned int*)&Bs[buf][wn * 32 + ni * 16 + l][0];
#pragma unroll
                for (int r = 0; r < 8; ++r) bfr[ni].u[r] = Br[8 * hl + r];
            } else {
                const unsigned short* BsU = (const unsigned short*)&Bs[buf][0][0];
                int col = wn * 32 + ni * 16 + l;
#pragma unroll
                for (int r = 0; r < 8; ++r) {
                    unsigned int lo = BsU[(16 * hl + 2 * r) * LDBN_S + col];
                    unsigned int hi = BsU[(16 * hl + 2 * r + 1) * LDBN_S + col];
                    bfr[ni].u[r] = lo | (hi << 16);
                }
            }
        }
        // ---- 8 WMMAs per wave per K tile
#pragma unroll
        for (int mi = 0; mi < 4; ++mi)
#pragma unroll
            for (int ni = 0; ni < 2; ++ni)
                acc[mi][ni] = __builtin_amdgcn_wmma_f32_16x16x32_bf16(
                    false, af[mi].v, false, bfr[ni].v,
                    (short)0, acc[mi][ni], false, false);

        if (!hasNext) break;

        // ---- stage next tile into the other buffer; single barrier per iter
        const int nb = buf ^ 1;
        *(uint4*)&As[nb][ra][sa * 8]      = a0;
        *(uint4*)&As[nb][ra + 64][sa * 8] = a1;
        if (BT) {
            *(uint4*)&Bs[nb][ra][sa * 8]      = b0;
            *(uint4*)&Bs[nb][ra + 64][sa * 8] = b1;
        } else {
            *(uint4*)&Bs[nb][rb][sb * 8]      = b0;
            *(uint4*)&Bs[nb][rb + 16][sb * 8] = b1;
        }
        __syncthreads();
        buf = nb;
    }

    // ---- epilogue: C/D layout VGPR r -> row (r + 8*half), col = lane&15
#pragma unroll
    for (int mi = 0; mi < 4; ++mi) {
#pragma unroll
        for (int ni = 0; ni < 2; ++ni) {
            int col  = bn + wn * 32 + ni * 16 + l;
            float bv = bias ? bias[col] : 0.0f;
#pragma unroll
            for (int r = 0; r < 8; ++r) {
                int row = bm + wm * 64 + mi * 16 + 8 * hl + r;
                float v = acc[mi][ni][r] * alpha + bv;
                long idx = z * sCz + (long)row * ldc + col;
                if (Cf) Cf[idx] = v;
                else    Ch[idx] = (bf16)v;
            }
        }
    }
}

// ---------------------------------------------------------------------------
// f32 -> bf16 conversion
// ---------------------------------------------------------------------------
__global__ __launch_bounds__(256)
void cvt_f32_to_bf16(const float* __restrict__ in, bf16* __restrict__ out, long n) {
    long i = (long)blockIdx.x * 256 + threadIdx.x;
    if (i < n) out[i] = (bf16)in[i];
}

// ---------------------------------------------------------------------------
// add sinusoidal positional encoding + convert to bf16; layout [B*T, 512]
// ---------------------------------------------------------------------------
__global__ __launch_bounds__(256)
void add_pe_cvt(const float* __restrict__ xp, bf16* __restrict__ xpb) {
    long i  = (long)blockIdx.x * 256 + threadIdx.x;
    int col = (int)(i & 511);
    int row = (int)(i >> 9);
    int t   = row & 511;                       // position within sequence
    // div = exp(-(ln 10000 / 512) * (2i)), 2i = col & ~1
    float div = __expf(-0.0179889025f * (float)(col & ~1));
    float ang = (float)t * div;
    float pe  = (col & 1) ? __cosf(ang) : __sinf(ang);
    xpb[i] = (bf16)(xp[i] + pe);
}

// ---------------------------------------------------------------------------
// row softmax: f32 [rows,512] -> bf16 [rows,512]; 256 threads, 2 cols each
// ---------------------------------------------------------------------------
__global__ __launch_bounds__(256)
void softmax_rows_bf16(const float* __restrict__ S, bf16* __restrict__ O) {
    const int row = blockIdx.x;
    const int t   = threadIdx.x;
    const float* r = S + (long)row * 512;
    float a = r[t], b = r[t + 256];
    __shared__ float red[256];
    __shared__ float bcast;
    red[t] = fmaxf(a, b);
    __syncthreads();
    for (int s = 128; s > 0; s >>= 1) {
        if (t < s) red[t] = fmaxf(red[t], red[t + s]);
        __syncthreads();
    }
    if (t == 0) bcast = red[0];
    __syncthreads();
    float m  = bcast;
    float e0 = __expf(a - m), e1 = __expf(b - m);
    __syncthreads();
    red[t] = e0 + e1;
    __syncthreads();
    for (int s = 128; s > 0; s >>= 1) {
        if (t < s) red[t] += red[t + s];
        __syncthreads();
    }
    if (t == 0) bcast = red[0];
    __syncthreads();
    float inv = 1.0f / bcast;
    O[(long)row * 512 + t]       = (bf16)(e0 * inv);
    O[(long)row * 512 + t + 256] = (bf16)(e1 * inv);
}

// ---------------------------------------------------------------------------
// attention pooling: per batch, scores = softmax(out @ query), pooled = p^T out
// out: [B*T, 512] f32, query: [512], pooled: [B, 512]
// ---------------------------------------------------------------------------
__global__ __launch_bounds__(512)
void pool_kernel(const float* __restrict__ out, const float* __restrict__ query,
                 float* __restrict__ pooled) {
    const int b = blockIdx.x;
    const int t = threadIdx.x;              // 0..511
    __shared__ float qs[512];
    __shared__ float p[512];
    __shared__ float red[512];
    __shared__ float bcast;
    qs[t] = query[t];
    __syncthreads();
    const float* row = out + ((long)b * 512 + t) * 512;
    float dot = 0.0f;
    for (int e = 0; e < 512; ++e) dot += row[e] * qs[e];
    red[t] = dot;
    __syncthreads();
    for (int s = 256; s > 0; s >>= 1) {
        if (t < s) red[t] = fmaxf(red[t], red[t + s]);
        __syncthreads();
    }
    if (t == 0) bcast = red[0];
    __syncthreads();
    float m  = bcast;
    float ex = __expf(dot - m);
    __syncthreads();
    red[t] = ex;
    __syncthreads();
    for (int s = 256; s > 0; s >>= 1) {
        if (t < s) red[t] += red[t + s];
        __syncthreads();
    }
    if (t == 0) bcast = red[0];
    __syncthreads();
    p[t] = ex / bcast;
    __syncthreads();
    float acc = 0.0f;
    for (int tt = 0; tt < 512; ++tt)
        acc += p[tt] * out[((long)b * 512 + tt) * 512 + t];
    pooled[(long)b * 512 + t] = acc;
}

// ---------------------------------------------------------------------------
// Host orchestration
// ---------------------------------------------------------------------------
extern "C" void kernel_launch(void* const* d_in, const int* in_sizes, int n_in,
                              void* d_out, int out_size, void* d_ws, size_t ws_size,
                              hipStream_t stream) {
    (void)in_sizes; (void)n_in; (void)out_size; (void)ws_size;
    // setup_inputs dict order: x, W_proj, W_k, W_q, W_v, W_unify, b_unify, query
    const float* x       = (const float*)d_in[0];
    const float* W_proj  = (const float*)d_in[1];
    const float* W_k     = (const float*)d_in[2];
    const float* W_q     = (const float*)d_in[3];
    const float* W_v     = (const float*)d_in[4];
    const float* W_unify = (const float*)d_in[5];
    const float* b_unify = (const float*)d_in[6];
    const float* query   = (const float*)d_in[7];
    float* pooled = (float*)d_out;

    const int B = 16, T = 512, E = 512, H = 8, HE = 4096;
    const long Mall = (long)B * T;          // 8192

    char* p = (char*)d_ws;
    auto carve = [&](size_t bytes) -> char* {
        char* r = p;
        p += (bytes + 255) & ~(size_t)255;
        return r;
    };
    bf16*  xb   = (bf16*)carve((size_t)Mall * E * 2);     // x in bf16
    bf16*  Wpb  = (bf16*)carve((size_t)E * E * 2);
    bf16*  Wqb  = (bf16*)carve((size_t)E * HE * 2);
    bf16*  Wkb  = (bf16*)carve((size_t)E * HE * 2);
    bf16*  Wvb  = (bf16*)carve((size_t)E * HE * 2);
    bf16*  Wub  = (bf16*)carve((size_t)HE * E * 2);
    float* xpf  = (float*)carve((size_t)Mall * E * 4);    // proj output f32
    bf16*  xpb  = (bf16*)carve((size_t)Mall * E * 2);     // proj + PE, bf16
    bf16*  qb   = (bf16*)carve((size_t)T * HE * 2);       // per-batch q
    bf16*  kb   = (bf16*)carve((size_t)T * HE * 2);       // per-batch k
    bf16*  vb   = (bf16*)carve((size_t)T * HE * 2);       // per-batch v
    float* sc   = (float*)carve((size_t)H * T * T * 4);   // per-batch scores f32
    bf16*  attb = (bf16*)carve((size_t)H * T * T * 2);    // per-batch softmax bf16
    bf16*  aout = (bf16*)carve((size_t)Mall * HE * 2);    // attention out [B*T, H*E]
    float* outf = (float*)carve((size_t)Mall * E * 4);    // unify output f32

    auto cvt = [&](const float* src, bf16* dst, long n) {
        cvt_f32_to_bf16<<<dim3((unsigned)((n + 255) / 256)), dim3(256), 0, stream>>>(src, dst, n);
    };
    cvt(x,       xb,  Mall * E);
    cvt(W_proj,  Wpb, (long)E * E);
    cvt(W_q,     Wqb, (long)E * HE);
    cvt(W_k,     Wkb, (long)E * HE);
    cvt(W_v,     Wvb, (long)E * HE);
    cvt(W_unify, Wub, (long)HE * E);

    // x @ W_proj -> f32
    gemm_wmma<false><<<dim3(E / BN, (unsigned)(Mall / BM), 1), 256, 0, stream>>>(
        xb, E, 0, Wpb, E, 0, xpf, nullptr, E, 0, E, 1.0f, nullptr);
    // + positional encoding, -> bf16
    add_pe_cvt<<<dim3((unsigned)(Mall * E / 256)), 256, 0, stream>>>(xpf, xpb);

    const float scale = 0.21022410381342863f;   // 512^(-1/4)
    for (int b = 0; b < B; ++b) {
        const bf16* xpB = xpb + (long)b * T * E;
        // q/k/v projections for this batch: [T,E] @ [E,HE] -> bf16 [T,HE]
        gemm_wmma<false><<<dim3(HE / BN, T / BM, 1), 256, 0, stream>>>(
            xpB, E, 0, Wqb, HE, 0, nullptr, qb, HE, 0, E, scale, nullptr);
        gemm_wmma<false><<<dim3(HE / BN, T / BM, 1), 256, 0, stream>>>(
            xpB, E, 0, Wkb, HE, 0, nullptr, kb, HE, 0, E, scale, nullptr);
        gemm_wmma<false><<<dim3(HE / BN, T / BM, 1), 256, 0, stream>>>(
            xpB, E, 0, Wvb, HE, 0, nullptr, vb, HE, 0, E, 1.0f, nullptr);
        // scores[h] = q_h @ k_h^T  -> f32 [T,T] per head (z = head)
        gemm_wmma<true><<<dim3(T / BN, T / BM, H), 256, 0, stream>>>(
            qb, HE, (long)E, kb, HE, (long)E, sc, nullptr, T, (long)T * T,
            E, 1.0f, nullptr);
        // softmax over last dim -> bf16
        softmax_rows_bf16<<<dim3(H * T), 256, 0, stream>>>(sc, attb);
        // out[h] = att_h @ v_h, written strided into [T, H*E] layout
        gemm_wmma<false><<<dim3(E / BN, T / BM, H), 256, 0, stream>>>(
            attb, T, (long)T * T, vb, HE, (long)E, nullptr,
            aout + (long)b * T * HE, HE, (long)E, T, 1.0f, nullptr);
    }

    // unify: [B*T, H*E] @ [H*E, E] + b_unify -> f32
    gemm_wmma<false><<<dim3(E / BN, (unsigned)(Mall / BM), 1), 256, 0, stream>>>(
        aout, HE, 0, Wub, E, 0, outf, nullptr, E, 0, HE, 1.0f, b_unify);

    // attention pooling -> d_out [B, E]
    pool_kernel<<<dim3(B), 512, 0, stream>>>(outf, query, pooled);
}
